// Attention_78323023610125
// MI455X (gfx1250) — compile-verified, matrix operand order checked by reading, not run on hardware
//
#include <hip/hip_runtime.h>

// ---------------- problem constants ----------------
constexpr int kB   = 2;
constexpr int kS   = 2048;
constexpr int kHID = 2048;
constexpr int kH   = 16;
constexpr int kHKV = 4;
constexpr int kD   = 128;
constexpr int kG   = kH / kHKV;          // 4
constexpr int kM   = kB * kS;            // 4096 tokens
constexpr int kNQKV = (kH + 2 * kHKV) * kD; // 3072
constexpr float kCLIP = 8.0f;
constexpr float kSCALE = 0.08838834764831845f;     // 1/sqrt(128)
constexpr float kLOGTH_OVER_HALF = 9.210340371976184f / 64.0f; // ln(10000)/(D/2)

typedef __attribute__((ext_vector_type(16))) __bf16 v16bf;
typedef __attribute__((ext_vector_type(8)))  float  v8f;
typedef int v4i __attribute__((vector_size(16)));
typedef __attribute__((address_space(1))) v4i* v4i_g;   // global ptr to b128
typedef __attribute__((address_space(3))) v4i* v4i_l;   // LDS ptr to b128

union Frag { unsigned u[8]; v16bf v; };

// Async global->LDS staging (gfx1250 GLOBAL_LOAD_ASYNC_TO_LDS_B128, ASYNCcnt).
#if defined(__gfx1250__) && \
    __has_builtin(__builtin_amdgcn_global_load_async_to_lds_b128) && \
    __has_builtin(__builtin_amdgcn_s_wait_asynccnt)
#define USE_ASYNC_LDS 1
#else
#define USE_ASYNC_LDS 0
#endif

__device__ inline v8f v8f_zero() {
  v8f z;
#pragma unroll
  for (int i = 0; i < 8; ++i) z[i] = 0.0f;
  return z;
}

__device__ inline v8f wmma_bf16(const Frag& a, const Frag& b, v8f c) {
  return __builtin_amdgcn_wmma_f32_16x16x32_bf16(
      false, a.v, false, b.v, (short)0, c, false, false);
}

// WMMA A/B fragment K-column for dword slot p (pairs of bf16), lane half h.
__device__ inline int frag_kcol(int p, int h) {
  return (p < 4) ? (8 * h + 2 * p) : (16 + 8 * h + 2 * (p - 4));
}

// ---------------- f32 -> bf16 convert ----------------
__global__ void __launch_bounds__(256) cvt_bf16_kernel(
    const float* __restrict__ in, __bf16* __restrict__ out, int n) {
  int i = blockIdx.x * blockDim.x + threadIdx.x;
  if (i < n) out[i] = (__bf16)in[i];
}

// ---------------- WMMA GEMM: Y[M,N] = A[M,K] * W[N,K]^T ----------------
// block = 256 threads (8 waves), tile 128(M) x 128(N), K step 64.
// -> 2x4 C tiles per wave, 16 WMMAs per barrier period.
template <bool CLIP_TO_BF16>
__global__ void __launch_bounds__(256) gemm_bf16_kernel(
    const __bf16* __restrict__ A, const __bf16* __restrict__ W,
    void* __restrict__ Y, int M, int N, int K) {
  __shared__ __align__(16) __bf16 As[128][64];
  __shared__ __align__(16) __bf16 Ws[128][64];

  const int tid  = threadIdx.x;
  const int wave = tid >> 5;
  const int lane = tid & 31;
  const int l16  = lane & 15;
  const int half = lane >> 4;
  const int mw   = (wave & 3) * 32;
  const int nw   = (wave >> 2) * 64;
  const int rowBase = blockIdx.y * 128;
  const int colBase = blockIdx.x * 128;

  v8f acc[2][4];
#pragma unroll
  for (int mi = 0; mi < 2; ++mi)
#pragma unroll
    for (int j = 0; j < 4; ++j) acc[mi][j] = v8f_zero();

  for (int k0 = 0; k0 < K; k0 += 64) {
    // ---- stage two 128x64 bf16 tiles (16 KB each); 8 x 16B chunks/thread ----
#pragma unroll
    for (int it = 0; it < 4; ++it) {
      int idx = tid + 256 * it;
      int r = idx >> 3;
      int c = (idx & 7) * 8;
      const __bf16* ga = A + (size_t)(rowBase + r) * K + k0 + c;
      const __bf16* gw = W + (size_t)(colBase + r) * K + k0 + c;
#if USE_ASYNC_LDS
      __builtin_amdgcn_global_load_async_to_lds_b128(
          (v4i_g)ga, (v4i_l)&As[r][c], 0, 0);
      __builtin_amdgcn_global_load_async_to_lds_b128(
          (v4i_g)gw, (v4i_l)&Ws[r][c], 0, 0);
#else
      *(uint4*)&As[r][c] = *(const uint4*)ga;
      *(uint4*)&Ws[r][c] = *(const uint4*)gw;
#endif
    }
#if USE_ASYNC_LDS
    __builtin_amdgcn_s_wait_asynccnt(0);
#endif
    __syncthreads();

    // ---- 2 K-chunks of 32, 16 WMMAs per wave ----
#pragma unroll
    for (int kc = 0; kc < 2; ++kc) {
      Frag a[2];
#pragma unroll
      for (int mi = 0; mi < 2; ++mi) {
        const unsigned* arow = (const unsigned*)&As[mw + 16 * mi + l16][32 * kc];
#pragma unroll
        for (int p = 0; p < 8; ++p) a[mi].u[p] = arow[frag_kcol(p, half) >> 1];
      }
#pragma unroll
      for (int j = 0; j < 4; ++j) {
        Frag bfr;
        const unsigned* wrow = (const unsigned*)&Ws[nw + 16 * j + l16][32 * kc];
#pragma unroll
        for (int p = 0; p < 8; ++p) bfr.u[p] = wrow[frag_kcol(p, half) >> 1];
#pragma unroll
        for (int mi = 0; mi < 2; ++mi)
          acc[mi][j] = wmma_bf16(a[mi], bfr, acc[mi][j]);
      }
    }
    __syncthreads();
  }

#pragma unroll
  for (int mi = 0; mi < 2; ++mi) {
#pragma unroll
    for (int j = 0; j < 4; ++j) {
#pragma unroll
      for (int r = 0; r < 8; ++r) {
        int row = rowBase + mw + 16 * mi + r + 8 * half;
        int col = colBase + nw + 16 * j + l16;
        float v = acc[mi][j][r];
        if (CLIP_TO_BF16) {
          v = fminf(fmaxf(v, -kCLIP), kCLIP);
          ((__bf16*)Y)[(size_t)row * N + col] = (__bf16)v;
        } else {
          ((float*)Y)[(size_t)row * N + col] = v;
        }
      }
    }
  }
}

// ---------------- RoPE + split/layout kernel ----------------
__global__ void __launch_bounds__(256) rope_split_kernel(
    const __bf16* __restrict__ qkv, const int* __restrict__ pos_ids,
    __bf16* __restrict__ Q, __bf16* __restrict__ Kh, __bf16* __restrict__ Vt) {
  const int bs = blockIdx.x;
  const int b = bs / kS, s = bs % kS;
  const float pos = (float)pos_ids[bs];
  const __bf16* x = qkv + (size_t)bs * kNQKV;

  for (int idx = threadIdx.x; idx < kH * (kD / 2); idx += blockDim.x) {
    int h = idx >> 6, d = idx & 63;
    float x1 = (float)x[h * kD + d];
    float x2 = (float)x[h * kD + d + 64];
    float ang = pos * __expf(-(float)d * kLOGTH_OVER_HALF);
    float sn, cs; __sincosf(ang, &sn, &cs);
    __bf16* qo = Q + (((size_t)b * kH + h) * kS + s) * kD;
    qo[d]      = (__bf16)((x1 * cs - x2 * sn) * kSCALE);
    qo[d + 64] = (__bf16)((x2 * cs + x1 * sn) * kSCALE);
  }
  for (int idx = threadIdx.x; idx < kHKV * (kD / 2); idx += blockDim.x) {
    int h = idx >> 6, d = idx & 63;
    float x1 = (float)x[kH * kD + h * kD + d];
    float x2 = (float)x[kH * kD + h * kD + d + 64];
    float ang = pos * __expf(-(float)d * kLOGTH_OVER_HALF);
    float sn, cs; __sincosf(ang, &sn, &cs);
    __bf16* ko = Kh + (((size_t)b * kHKV + h) * kS + s) * kD;
    ko[d]      = (__bf16)(x1 * cs - x2 * sn);
    ko[d + 64] = (__bf16)(x2 * cs + x1 * sn);
  }
  for (int idx = threadIdx.x; idx < kHKV * kD; idx += blockDim.x) {
    int h = idx >> 7, d = idx & 127;
    Vt[(((size_t)b * kHKV + h) * kD + d) * kS + s] = x[(kH + kHKV) * kD + idx];
  }
}

// ---------------- flash attention: one wave per (b, head, 16-query tile) ----
__global__ void __launch_bounds__(32) attn_kernel(
    const __bf16* __restrict__ Q, const __bf16* __restrict__ Kh,
    const __bf16* __restrict__ Vt, const int* __restrict__ seq_ids,
    __bf16* __restrict__ AO) {
  const int qt  = blockIdx.x;
  const int h   = blockIdx.y;
  const int b   = blockIdx.z;
  const int hkv = h / kG;
  const int lane = threadIdx.x;
  const int l16  = lane & 15;
  const int half = lane >> 4;

  __shared__ __align__(16) __bf16 Plds[16][32];

  const __bf16* qbase = Q + (((size_t)b * kH + h) * kS + (size_t)qt * 16) * kD;
  Frag qf[4];
#pragma unroll
  for (int c = 0; c < 4; ++c) {
    const unsigned* qrow = (const unsigned*)(qbase + (size_t)l16 * kD + 32 * c);
#pragma unroll
    for (int p = 0; p < 8; ++p) qf[c].u[p] = qrow[frag_kcol(p, half) >> 1];
  }

  const int* seqb = seq_ids + (size_t)b * kS;
  int sq[8];
#pragma unroll
  for (int r = 0; r < 8; ++r) sq[r] = seqb[qt * 16 + r + 8 * half];

  float mrow[8], lrow[8];
  v8f oacc[8];
#pragma unroll
  for (int r = 0; r < 8; ++r) { mrow[r] = -1e30f; lrow[r] = 0.0f; }
#pragma unroll
  for (int j = 0; j < 8; ++j) oacc[j] = v8f_zero();

  const __bf16* kbase = Kh + ((size_t)b * kHKV + hkv) * (size_t)kS * kD;
  const __bf16* vbase = Vt + ((size_t)b * kHKV + hkv) * (size_t)kD * kS;
  const int kend = qt * 16 + 16;

  for (int kb = 0; kb < kend; kb += 32) {
    // ---- S = Q * K^T : 2 key subtiles x 4 D-chunks = 8 WMMAs ----
    v8f sacc[2];
    sacc[0] = v8f_zero(); sacc[1] = v8f_zero();
#pragma unroll
    for (int t = 0; t < 2; ++t) {
      const unsigned* krow =
          (const unsigned*)(kbase + (size_t)(kb + 16 * t + l16) * kD);
#pragma unroll
      for (int c = 0; c < 4; ++c) {
        Frag kf;
#pragma unroll
        for (int p = 0; p < 8; ++p)
          kf.u[p] = krow[(32 * c + frag_kcol(p, half)) >> 1];
        sacc[t] = wmma_bf16(qf[c], kf, sacc[t]);
      }
    }

    // ---- mask + online softmax ----
    const int sk0 = seqb[kb + l16];
    const int sk1 = seqb[kb + 16 + l16];
#pragma unroll
    for (int r = 0; r < 8; ++r) {
      const int m = r + 8 * half;
      const int qidx = qt * 16 + m;
      float s0 = sacc[0][r];
      float s1 = sacc[1][r];
      const bool v0 = (qidx >= kb + l16)      && (sq[r] == sk0);
      const bool v1 = (qidx >= kb + 16 + l16) && (sq[r] == sk1);
      s0 = v0 ? s0 : -1e30f;
      s1 = v1 ? s1 : -1e30f;
      float bm = fmaxf(s0, s1);
#pragma unroll
      for (int off = 1; off < 16; off <<= 1) bm = fmaxf(bm, __shfl_xor(bm, off, 16));
      const float mnew  = fmaxf(mrow[r], bm);
      const float alpha = __expf(mrow[r] - mnew);
      const float p0 = v0 ? __expf(s0 - mnew) : 0.0f;
      const float p1 = v1 ? __expf(s1 - mnew) : 0.0f;
      float ps = p0 + p1;
#pragma unroll
      for (int off = 1; off < 16; off <<= 1) ps += __shfl_xor(ps, off, 16);
      lrow[r] = lrow[r] * alpha + ps;
      mrow[r] = mnew;
#pragma unroll
      for (int j = 0; j < 8; ++j) oacc[j][r] *= alpha;
      Plds[m][l16]      = (__bf16)p0;
      Plds[m][16 + l16] = (__bf16)p1;
    }
    __syncthreads();

    // ---- O += P * V : one P A-fragment, 8 D-tiles = 8 WMMAs ----
    Frag pf;
    const unsigned* prow = (const unsigned*)&Plds[l16][0];
#pragma unroll
    for (int p = 0; p < 8; ++p) pf.u[p] = prow[frag_kcol(p, half) >> 1];
#pragma unroll
    for (int j = 0; j < 8; ++j) {
      Frag vf;
      const unsigned* vrow =
          (const unsigned*)(vbase + (size_t)(16 * j + l16) * kS + kb);
#pragma unroll
      for (int p = 0; p < 8; ++p) vf.u[p] = vrow[frag_kcol(p, half) >> 1];
      oacc[j] = wmma_bf16(pf, vf, oacc[j]);
    }
    __syncthreads();
  }

  __bf16* aobase = AO + ((size_t)b * kS + (size_t)qt * 16) * (kH * kD) + h * kD;
#pragma unroll
  for (int r = 0; r < 8; ++r) {
    const int m = r + 8 * half;
    const float inv = 1.0f / lrow[r];
#pragma unroll
    for (int j = 0; j < 8; ++j)
      aobase[(size_t)m * (kH * kD) + 16 * j + l16] = (__bf16)(oacc[j][r] * inv);
  }
}

// ---------------- host launch ----------------
extern "C" void kernel_launch(void* const* d_in, const int* in_sizes, int n_in,
                              void* d_out, int out_size, void* d_ws, size_t ws_size,
                              hipStream_t stream) {
  (void)in_sizes; (void)n_in; (void)out_size; (void)ws_size;
  const float* hidden = (const float*)d_in[0];
  const int*   wpos   = (const int*)d_in[1];   // within_seq_position_ids
  /* d_in[2] = global_position_ids (unused by this layer) */
  const int*   seq    = (const int*)d_in[3];
  const float* Wq     = (const float*)d_in[4];
  const float* Wk     = (const float*)d_in[5];
  const float* Wv     = (const float*)d_in[6];
  const float* Wo     = (const float*)d_in[7];

  __bf16* p = (__bf16*)d_ws;
  __bf16* X16    = p; p += (size_t)kM * kHID;
  __bf16* Wqkv16 = p; p += (size_t)kNQKV * kHID;
  __bf16* Wo16   = p; p += (size_t)kHID * (kH * kD);
  __bf16* QKV16  = p; p += (size_t)kM * kNQKV;
  __bf16* Q16    = p; p += (size_t)kB * kH * kS * kD;
  __bf16* K16    = p; p += (size_t)kB * kHKV * kS * kD;
  __bf16* V16t   = p; p += (size_t)kB * kHKV * kD * kS;
  __bf16* AO16   = p; p += (size_t)kM * (kH * kD);

  auto cvt = [&](const float* src, __bf16* dst, int n) {
    cvt_bf16_kernel<<<(n + 255) / 256, 256, 0, stream>>>(src, dst, n);
  };
  cvt(hidden, X16, kM * kHID);
  cvt(Wq, Wqkv16,                                    kH * kD * kHID);
  cvt(Wk, Wqkv16 + (size_t)kH * kD * kHID,           kHKV * kD * kHID);
  cvt(Wv, Wqkv16 + (size_t)(kH + kHKV) * kD * kHID,  kHKV * kD * kHID);
  cvt(Wo, Wo16, kHID * kH * kD);

  // QKV projection (clip fused, bf16 out): tiles 128x128
  gemm_bf16_kernel<true><<<dim3(kNQKV / 128, kM / 128), 256, 0, stream>>>(
      X16, Wqkv16, QKV16, kM, kNQKV, kHID);

  rope_split_kernel<<<kB * kS, 256, 0, stream>>>(QKV16, wpos, Q16, K16, V16t);

  attn_kernel<<<dim3(kS / 16, kH, kB), 32, 0, stream>>>(Q16, K16, V16t, seq, AO16);

  gemm_bf16_kernel<false><<<dim3(kHID / 128, kM / 128), 256, 0, stream>>>(
      AO16, Wo16, d_out, kM, kHID, kH * kD);
}
